// BaselineProjCE_2542620639592
// MI455X (gfx1250) — compile-verified, matrix operand order checked by reading, not run on hardware
//
#include <hip/hip_runtime.h>
#include <math.h>

// Problem constants (B=2, S=2048, H=2048, V=128000)
#define H      2048
#define V      128000
#define NROWS  4096          // B*S
#define BM     128           // rows per block
#define BN     128           // vocab cols per block
#define BK     64            // K section staged in LDS
#define BLOCK  256           // 8 wave32
#define LDS_STRIDE 72        // halves per row (64 + 8 pad) -> conflict-free b128 reads

typedef __attribute__((ext_vector_type(16))) __bf16        v16bf;
typedef __attribute__((ext_vector_type(8)))  float         v8f;
typedef __attribute__((ext_vector_type(4)))  float         fv4;
typedef __attribute__((ext_vector_type(4)))  unsigned int  uv4;

union BFrag {
    v16bf v;
    uv4   q[2];
};

// Pack two f32 -> packed bf16 (truncation) in ONE v_perm_b32.
// result = (bf16(hi) << 16) | bf16(lo)
__device__ __forceinline__ unsigned int pack2_bf16(float lo, float hi) {
    return __builtin_amdgcn_perm(__float_as_uint(hi), __float_as_uint(lo), 0x07060302u);
}

// ---------------------------------------------------------------------------
// Kernel 1: exact f32 target logit  t[n] = dot(x[n,:], W[y[n],:])
// ---------------------------------------------------------------------------
__global__ __launch_bounds__(BLOCK) void tlogit_kernel(
    const float* __restrict__ x, const int* __restrict__ y,
    const float* __restrict__ W, float* __restrict__ tlogit)
{
    __shared__ float red[BLOCK / 32];
    const int n   = blockIdx.x;
    const int tid = threadIdx.x;
    const int t   = y[n];
    const float* xr = x + (size_t)n * H;
    const float* wr = W + (size_t)t * H;

    float acc = 0.0f;
    for (int h = tid; h < H; h += BLOCK) acc = fmaf(xr[h], wr[h], acc);
    #pragma unroll
    for (int off = 16; off > 0; off >>= 1) acc += __shfl_xor(acc, off, 32);
    if ((tid & 31) == 0) red[tid >> 5] = acc;
    __syncthreads();
    if (tid == 0) {
        float a = 0.0f;
        #pragma unroll
        for (int w = 0; w < BLOCK / 32; ++w) a += red[w];
        tlogit[n] = a;
    }
}

// ---------------------------------------------------------------------------
// Kernel 2: fused bf16-WMMA GEMM + exp(logit - t_row) partial row sums.
// Grid: (V/BN, NROWS/BM). Block: 256 thr = 8 waves, wave = 32x64 output patch.
// ---------------------------------------------------------------------------
__global__ __launch_bounds__(BLOCK) void gemm_expsum_kernel(
    const float* __restrict__ x, const float* __restrict__ W,
    const float* __restrict__ tlogit, float* __restrict__ ssum)
{
    __shared__ unsigned short lA[BM * LDS_STRIDE];   // x tile, bf16
    __shared__ unsigned short lB[BN * LDS_STRIDE];   // W tile, bf16

    const int tid  = threadIdx.x;
    const int wave = tid >> 5;
    const int lane = tid & 31;
    const int m15  = lane & 15;
    const int hi   = lane >> 4;          // 0: lanes 0-15, 1: lanes 16-31
    const int rg   = wave >> 1;          // row group 0..3  (32 rows each)
    const int cg   = wave & 1;           // col group 0..1  (64 cols each)

    const int rowBase = blockIdx.y * BM;
    const int colBase = blockIdx.x * BN;

    // Staging assignment for this thread (fixed across sections):
    // one 16-byte bf16 block = 8 consecutive f32 source elements per iteration.
    const int sr0  = tid >> 3;           // row 0..31 handled in 4 strides of 32
    const int sblk = tid & 7;            // 16B block within the BK section

    // Per-wave LDS fragment base addresses (constant across the K loop).
    const unsigned short* paf[2];
    const unsigned short* pbf[4];
    #pragma unroll
    for (int rt = 0; rt < 2; ++rt)
        paf[rt] = &lA[(rg * 32 + rt * 16 + m15) * LDS_STRIDE + hi * 8];
    #pragma unroll
    for (int ct = 0; ct < 4; ++ct)
        pbf[ct] = &lB[(cg * 64 + ct * 16 + m15) * LDS_STRIDE + hi * 8];

    v8f c[2][4];
    #pragma unroll
    for (int rt = 0; rt < 2; ++rt)
        #pragma unroll
        for (int ct = 0; ct < 4; ++ct)
            c[rt][ct] = (v8f){0.f, 0.f, 0.f, 0.f, 0.f, 0.f, 0.f, 0.f};

    for (int ks = 0; ks < H; ks += BK) {
        __syncthreads();
        // Stage A (x) and B (W) 128xBK sections as bf16 into LDS,
        // and speculatively prefetch the NEXT section's cachelines
        // (global_prefetch_b8; dropped silently if the tail runs past W).
        #pragma unroll
        for (int s = 0; s < 4; ++s) {
            const int r = sr0 + s * 32;
            const float* sa = x + (size_t)(rowBase + r) * H + ks + sblk * 8;
            const float* sb = W + (size_t)(colBase + r) * H + ks + sblk * 8;
            if (ks + BK < H) {
                __builtin_prefetch(sa + BK, 0, 1);
                __builtin_prefetch(sb + BK, 0, 1);
            }
            fv4 a0 = *(const fv4*)sa, a1 = *(const fv4*)(sa + 4);
            fv4 b0 = *(const fv4*)sb, b1 = *(const fv4*)(sb + 4);
            uv4 pa = (uv4){pack2_bf16(a0.x, a0.y), pack2_bf16(a0.z, a0.w),
                           pack2_bf16(a1.x, a1.y), pack2_bf16(a1.z, a1.w)};
            uv4 pb = (uv4){pack2_bf16(b0.x, b0.y), pack2_bf16(b0.z, b0.w),
                           pack2_bf16(b1.x, b1.y), pack2_bf16(b1.z, b1.w)};
            *(uv4*)&lA[r * LDS_STRIDE + sblk * 8] = pa;
            *(uv4*)&lB[r * LDS_STRIDE + sblk * 8] = pb;
        }
        __syncthreads();

        // 2 WMMA K-steps of 32 per section; all 12 ds_load_b128 of a step are
        // issued before the 8 WMMAs so the dscnt waits stagger instead of
        // draining to 0 before every WMMA pair.
        #pragma unroll
        for (int k0 = 0; k0 < BK; k0 += 32) {
            BFrag a[2], b[4];
            #pragma unroll
            for (int rt = 0; rt < 2; ++rt) {
                a[rt].q[0] = *(const uv4*)(paf[rt] + k0);
                a[rt].q[1] = *(const uv4*)(paf[rt] + k0 + 16);
            }
            #pragma unroll
            for (int ct = 0; ct < 4; ++ct) {
                b[ct].q[0] = *(const uv4*)(pbf[ct] + k0);
                b[ct].q[1] = *(const uv4*)(pbf[ct] + k0 + 16);
            }
            #pragma unroll
            for (int ct = 0; ct < 4; ++ct) {
                c[0][ct] = __builtin_amdgcn_wmma_f32_16x16x32_bf16(
                    false, a[0].v, false, b[ct].v, (short)0, c[0][ct], false, false);
                c[1][ct] = __builtin_amdgcn_wmma_f32_16x16x32_bf16(
                    false, a[1].v, false, b[ct].v, (short)0, c[1][ct], false, false);
            }
        }
    }

    // Epilogue: per row, sum exp(logit - tlogit[row]) over this block's 128 cols.
    // C layout: VGPR j -> row j (lanes 0-15) / row 8+j (lanes 16-31), col = m15.
    #pragma unroll
    for (int rt = 0; rt < 2; ++rt) {
        #pragma unroll
        for (int j = 0; j < 8; ++j) {
            const int grow = rowBase + rg * 32 + rt * 16 + hi * 8 + j;
            const float tl = tlogit[grow];
            float v = __expf(c[rt][0][j] - tl) + __expf(c[rt][1][j] - tl) +
                      __expf(c[rt][2][j] - tl) + __expf(c[rt][3][j] - tl);
            #pragma unroll
            for (int off = 1; off < 16; off <<= 1) v += __shfl_xor(v, off, 32);
            if (m15 == 0) atomicAdd(&ssum[grow], v);
        }
    }
}

// ---------------------------------------------------------------------------
// Kernel 3: loss_n = lse_n - t_n = log(ssum[n]); mean -> scalar out
// ---------------------------------------------------------------------------
__global__ __launch_bounds__(BLOCK) void finalize_kernel(
    const float* __restrict__ ssum, float* __restrict__ out)
{
    const int n = blockIdx.x * BLOCK + threadIdx.x;
    if (n < NROWS) atomicAdd(out, __logf(ssum[n]) * (1.0f / NROWS));
}

// ---------------------------------------------------------------------------
extern "C" void kernel_launch(void* const* d_in, const int* in_sizes, int n_in,
                              void* d_out, int out_size, void* d_ws, size_t ws_size,
                              hipStream_t stream) {
    const float* x = (const float*)d_in[0];
    const int*   y = (const int*)d_in[1];
    const float* W = (const float*)d_in[2];
    float* out = (float*)d_out;

    float* ssum   = (float*)d_ws;            // NROWS floats
    float* tlogit = ssum + NROWS;            // NROWS floats

    hipMemsetAsync(out,  0, sizeof(float), stream);
    hipMemsetAsync(ssum, 0, NROWS * sizeof(float), stream);

    tlogit_kernel<<<NROWS, BLOCK, 0, stream>>>(x, y, W, tlogit);

    dim3 grid(V / BN, NROWS / BM);           // (1000, 32)
    gemm_expsum_kernel<<<grid, BLOCK, 0, stream>>>(x, W, tlogit, ssum);

    finalize_kernel<<<(NROWS + BLOCK - 1) / BLOCK, BLOCK, 0, stream>>>(ssum, out);
}
